// PAM_Module_15522011807788
// MI455X (gfx1250) — matrix operand order repaired and checked by reading, not compiled, not run on hardware
//
#include <hip/hip_runtime.h>

#define NUM 2
#define BB  8
#define CC  512
#define HWN 1024   // H*W = 32*32

typedef __attribute__((ext_vector_type(16))) __bf16 v16bf;
typedef __attribute__((ext_vector_type(8)))  __bf16 v8bf;
typedef __attribute__((ext_vector_type(8)))  float  v8f;

__device__ __forceinline__ v8f wmma_bf16(v16bf a, v16bf b, v8f c) {
  return __builtin_amdgcn_wmma_f32_16x16x32_bf16(false, a, false, b, (short)0, c,
                                                 false, false);
}

// A-operand 16x32 bf16 (M=16 rows, K=32): lane L -> row n=L%16, group g=L/16,
// elements e<8: K = 8g+e ; e>=8: K = 16+8g+(e-8). Two 16B loads per lane.
__device__ __forceinline__ v16bf load_a16x32(const __bf16* base, int rowstride,
                                             int lane, int k0) {
  const int n = lane & 15, g = lane >> 4;
  const __bf16* p = base + (size_t)n * rowstride + k0 + 8 * g;
  v8bf r0 = *(const v8bf*)p;
  v8bf r1 = *(const v8bf*)(p + 16);
  v16bf a;
#pragma unroll
  for (int e = 0; e < 8; ++e) { a[e] = r0[e]; a[e + 8] = r1[e]; }
  return a;
}

// B-operand 32x16 bf16 (K=32 rows, N=16): lane L holds column N=L%16, g=L/16,
// elements e<8: K = 8g+e ; e>=8: K = 16+8g+(e-8). colptr = &B[K=0][N], K stride.
__device__ __forceinline__ v16bf load_b32x16(const __bf16* colptr, int stride, int g) {
  v16bf b;
#pragma unroll
  for (int e = 0; e < 8; ++e) {
    b[e] = colptr[(size_t)(8 * g + e) * stride];
    b[e + 8] = colptr[(size_t)(16 + 8 * g + e) * stride];
  }
  return b;
}

// gfx1250 async global->LDS copy (ASYNCcnt-tracked); one instr = 16B/lane.
__device__ __forceinline__ void async_g2l_b128(unsigned lds_byte_addr,
                                               const void* gptr) {
  asm volatile("global_load_async_to_lds_b128 %0, %1, off"
               :: "v"(lds_byte_addr), "v"(gptr) : "memory");
}
__device__ __forceinline__ void wait_async0() {
  asm volatile("s_wait_asynccnt 0" ::: "memory");
}

// ---------------------------------------------------------------------------
// Projection: out_T[o,p] = sum_c W[o,c] * X[c,p] + bias[o], f32 in, bf16 out.
// proj 0 (Q): X=tgt, out pixel-major (h,b,p,o)
// proj 1 (K): X=src, out chan-major  (h,b,o,m)
// proj 2 (V): X=tgt, out pixel-major (h,b,p,o)
// One wave per 16(o) x 16(p) tile; K-loop over c in steps of 32.
// ---------------------------------------------------------------------------
__global__ void proj_kernel(const float* __restrict__ fsrc,
                            const float* __restrict__ ftgt,
                            const float* __restrict__ Wq, const float* __restrict__ bq,
                            const float* __restrict__ Wk, const float* __restrict__ bk,
                            const float* __restrict__ Wv, const float* __restrict__ bv,
                            unsigned short* __restrict__ Qw,
                            unsigned short* __restrict__ Kw,
                            unsigned short* __restrict__ Vw) {
  const int lane = threadIdx.x & 31;
  int t = blockIdx.x * (blockDim.x >> 5) + (threadIdx.x >> 5);
  const int ptile = t & 63; t >>= 6;
  const int otile = t & 31; t >>= 5;
  const int b     = t & 7;  t >>= 3;
  const int h     = t & 1;  t >>= 1;
  const int proj  = t;
  if (proj > 2) return;

  const float *X, *Wm, *bias;
  if (proj == 0)      { X = ftgt; Wm = Wq; bias = bq; }
  else if (proj == 1) { X = fsrc; Wm = Wk; bias = bk; }
  else                { X = ftgt; Wm = Wv; bias = bv; }
  X    += (size_t)(h * BB + b) * CC * HWN;
  Wm   += (size_t)h * CC * CC;
  bias += h * CC;

  const int o0 = otile * 16, p0 = ptile * 16;
  const int n = lane & 15, g = lane >> 4;

  v8f acc = {};
#pragma unroll 2
  for (int c0 = 0; c0 < CC; c0 += 32) {
    // A tile: W rows (o0+n), f32 -> bf16
    const float* wp = Wm + (size_t)(o0 + n) * CC + c0 + 8 * g;
    v8f w0 = *(const v8f*)wp;
    v8f w1 = *(const v8f*)(wp + 16);
    v16bf a;
#pragma unroll
    for (int e = 0; e < 8; ++e) { a[e] = (__bf16)w0[e]; a[e + 8] = (__bf16)w1[e]; }
    // B tile: X column (p0+n), channels c0+8g+e / c0+16+8g+e, f32 -> bf16
    const float* xp = X + (size_t)c0 * HWN + p0 + n;
    v16bf bm;
#pragma unroll
    for (int e = 0; e < 8; ++e) { bm[e] = (__bf16)xp[(size_t)(8 * g + e) * HWN]; bm[e + 8] = (__bf16)xp[(size_t)(16 + 8 * g + e) * HWN]; }
    acc = wmma_bf16(a, bm, acc);
  }

  v8f bvec = *(const v8f*)(bias + o0 + 8 * g);
  if (proj == 1) {
    __bf16* Kb = reinterpret_cast<__bf16*>(Kw);
    const int m = p0 + n;
#pragma unroll
    for (int r = 0; r < 8; ++r) {
      const int o = o0 + 8 * g + r;
      Kb[((size_t)(h * BB + b) * CC + o) * HWN + m] = (__bf16)(acc[r] + bvec[r]);
    }
  } else {
    __bf16* Ob = reinterpret_cast<__bf16*>(proj == 0 ? Qw : Vw);
    const int p = p0 + n;
    v8bf q;
#pragma unroll
    for (int r = 0; r < 8; ++r) q[r] = (__bf16)(acc[r] + bvec[r]);
    *(v8bf*)(Ob + ((size_t)(h * BB + b) * HWN + p) * CC + o0 + 8 * g) = q;
  }
}

// ---------------------------------------------------------------------------
// Fused attention: one wave per (i, b, 16-row tile of n).
//   E_j[n,m] = Q[n,:]·K_j[:,m] ; attn = sum_j softmax_m(E_j)
//   out[c,n] = sum_m attn[n,m] * V[m,c]
// Q A-fragments are register-resident (16 x v16bf) for the whole kernel;
// Q tile is staged to LDS with async global->LDS loads first.
// ---------------------------------------------------------------------------
__global__ void __launch_bounds__(32)
attn_kernel(const unsigned short* __restrict__ Qw,
            const unsigned short* __restrict__ Kw,
            const unsigned short* __restrict__ Vw,
            float* __restrict__ out,
            float* __restrict__ partials) {
  __shared__ __align__(16) __bf16 Qs[16 * CC];     // 16 KB: Q row tile
  __shared__ float  attnF[16 * HWN];               // 64 KB: summed softmax (f32)
  __shared__ __align__(16) __bf16 attnB[16 * HWN]; // 32 KB: bf16 copy for WMMA
  __shared__ float  Es[16 * 32];                   //  2 KB: E chunk scratch
  __shared__ float  Mrow[16], Srow[16], Raw[16];

  const __bf16* Qb = reinterpret_cast<const __bf16*>(Qw);
  const __bf16* Kb = reinterpret_cast<const __bf16*>(Kw);
  const __bf16* Vb = reinterpret_cast<const __bf16*>(Vw);

  const int lane = threadIdx.x;
  int t = blockIdx.x;
  const int ntile = t & 63; t >>= 6;
  const int b = t & 7;      t >>= 3;
  const int i = t;
  const int n0 = ntile * 16;
  const int nloc = lane & 15, g = lane >> 4;

  // async-stage Q tile (rows n0..n0+15 contiguous, 16 KB = 32 b128/lane issues)
  {
    const char* src =
        (const char*)(Qb + ((size_t)(i * BB + b) * HWN + n0) * CC);
    const unsigned dst = (unsigned)(size_t)(void*)Qs;  // flat[31:0] == LDS addr
#pragma unroll
    for (int k = 0; k < 32; ++k) {
      const unsigned off = (unsigned)(k * 32 + lane) * 16u;
      async_g2l_b128(dst + off, src + off);
    }
    wait_async0();
  }
  __syncthreads();

  // Hoist Q A-fragments into registers: 16 chunks x v16bf = 128 VGPRs.
  v16bf aQ[16];
#pragma unroll
  for (int cc = 0; cc < 16; ++cc) aQ[cc] = load_a16x32(Qs, CC, lane, cc * 32);

  for (int j = 0; j < NUM; ++j) {
    const __bf16* Kbase = Kb + (size_t)(j * BB + b) * CC * HWN;
    float Mr = -3.0e38f, Sr = 0.0f, raw = 0.0f;

    for (int pass = 0; pass < 2; ++pass) {
      for (int mc = 0; mc < HWN / 32; ++mc) {
        const int m0 = mc * 32;
        v8f e0 = {}, e1 = {};
#pragma unroll
        for (int cc = 0; cc < 16; ++cc) {
          const __bf16* kr = Kbase + (size_t)(cc * 32) * HWN + m0 + nloc;
          e0 = wmma_bf16(aQ[cc], load_b32x16(kr, HWN, g), e0);
          e1 = wmma_bf16(aQ[cc], load_b32x16(kr + 16, HWN, g), e1);
        }
#pragma unroll
        for (int r = 0; r < 8; ++r) {
          Es[(r + 8 * g) * 32 + nloc]      = e0[r];
          Es[(r + 8 * g) * 32 + 16 + nloc] = e1[r];
        }
        __syncthreads();
        if (pass == 0) {
          if (lane < 16) {          // online row max / exp-sum
            float cmax = -3.0e38f;
            for (int k = 0; k < 32; ++k) cmax = fmaxf(cmax, Es[lane * 32 + k]);
            float newM = fmaxf(Mr, cmax);
            Sr *= __expf(Mr - newM);
            for (int k = 0; k < 32; ++k) Sr += __expf(Es[lane * 32 + k] - newM);
            Mr = newM;
          } else {                  // raw energy sum for alphas
            const int row = lane - 16;
            for (int k = 0; k < 32; ++k) raw += Es[row * 32 + k];
          }
        } else {                    // normalized softmax into attnF
          const float m = Mrow[nloc], sinv = 1.0f / Srow[nloc];
          const int cb = g * 16;
          for (int k = 0; k < 16; ++k) {
            float p = __expf(Es[nloc * 32 + cb + k] - m) * sinv;
            const int idx = nloc * HWN + m0 + cb + k;
            if (j == 0) attnF[idx] = p; else attnF[idx] += p;
          }
        }
        __syncthreads();
      }
      if (pass == 0) {
        if (lane < 16) { Mrow[lane] = Mr; Srow[lane] = Sr; }
        else           Raw[lane - 16] = raw;
        __syncthreads();
        if (lane == 0) {
          float s = 0.0f;
          for (int r = 0; r < 16; ++r) s += Raw[r];
          partials[(size_t)blockIdx.x * NUM + j] = s;  // deterministic partial
        }
        __syncthreads();
      }
    }
  }

  // f32 -> bf16 for the output GEMM's A operand
  for (int k = lane; k < 16 * HWN; k += 32) attnB[k] = (__bf16)attnF[k];
  __syncthreads();

  // out[n,c] = sum_m attn[n,m] * V[m,c]; C-layout rows are contiguous in n.
  const __bf16* Vbase = Vb + (size_t)(i * BB + b) * HWN * CC;
  float* obase = out + (size_t)(i * BB + b) * CC * HWN;
  for (int ct = 0; ct < CC / 16; ++ct) {
    const int c0 = ct * 16;
    v8f acc = {};
    for (int mc = 0; mc < HWN / 32; ++mc) {
      const int m0 = mc * 32;
      v16bf a  = load_a16x32(attnB, HWN, lane, m0);
      v16bf bm = load_b32x16(Vbase + (size_t)m0 * CC + c0 + nloc, CC, g);
      acc = wmma_bf16(a, bm, acc);
    }
    *(v8f*)(obase + (size_t)(c0 + nloc) * HWN + n0 + 8 * g) = acc;
  }
}

// means[i][j] = sum(E_ij)/(B*HW*HW); alphas = softmax_j(means)
__global__ void alphas_kernel(const float* __restrict__ partials,
                              float* __restrict__ alphas) {
  if (threadIdx.x != 0 || blockIdx.x != 0) return;
  const float inv = 1.0f / ((float)BB * (float)HWN * (float)HWN);
  for (int i = 0; i < NUM; ++i) {
    float m[NUM];
    for (int j = 0; j < NUM; ++j) {
      float s = 0.0f;
      for (int k = 0; k < BB * (HWN / 16); ++k)
        s += partials[(size_t)(i * BB * (HWN / 16) + k) * NUM + j];
      m[j] = s * inv;
    }
    float mx = fmaxf(m[0], m[1]);
    float e0 = __expf(m[0] - mx), e1 = __expf(m[1] - mx);
    float den = e0 + e1;
    alphas[i * NUM + 0] = e0 / den;
    alphas[i * NUM + 1] = e1 / den;
  }
}

extern "C" void kernel_launch(void* const* d_in, const int* in_sizes, int n_in,
                              void* d_out, int out_size, void* d_ws, size_t ws_size,
                              hipStream_t stream) {
  (void)in_sizes; (void)n_in; (void)out_size; (void)ws_size;
  const float* fsrc = (const float*)d_in[0];
  const float* ftgt = (const float*)d_in[1];
  const float* Wq = (const float*)d_in[2];
  const float* bq = (const float*)d_in[3];
  const float* Wk = (const float*)d_in[4];
  const float* bk = (const float*)d_in[5];
  const float* Wv = (const float*)d_in[6];
  const float* bv = (const float*)d_in[7];
  float* out = (float*)d_out;

  const size_t NPROJ = (size_t)NUM * BB * HWN * CC;  // 8,388,608 elements
  unsigned short* Qw = (unsigned short*)d_ws;
  unsigned short* Kw = Qw + NPROJ;
  unsigned short* Vw = Kw + NPROJ;
  float* partials = (float*)(Vw + NPROJ);            // 2048 floats

  // 3 proj * 2 heads * 8 batch * 32 o-tiles * 64 p-tiles = 98304 waves, 8/block
  proj_kernel<<<12288, 256, 0, stream>>>(fsrc, ftgt, Wq, bq, Wk, bk, Wv, bv,
                                         Qw, Kw, Vw);
  // num * B * (HW/16) = 1024 blocks, one wave each
  attn_kernel<<<1024, 32, 0, stream>>>(Qw, Kw, Vw, out, partials);
  alphas_kernel<<<1, 32, 0, stream>>>(partials, out + NPROJ);
}
